// SlotFeatureExtractor_78219944394834
// MI455X (gfx1250) — matrix operand order repaired and checked by reading, not compiled, not run on hardware
//
#include <hip/hip_runtime.h>
#include <hip/hip_bf16.h>
#include <stdint.h>

typedef __attribute__((ext_vector_type(16))) _Float16 v16h;
typedef __attribute__((ext_vector_type(8)))  float    v8f;

// ---------------------------------------------------------------- helpers ---
__device__ __forceinline__ float geluf(float x) {
    // tanh approximation (jax.nn.gelu default)
    float x3 = x * x * x;
    return 0.5f * x * (1.f + tanhf(0.7978845608028654f * (x + 0.044715f * x3)));
}
__device__ __forceinline__ float sigm(float x) { return 1.f / (1.f + expf(-x)); }
__device__ __forceinline__ float softplusf(float x) {
    return (x > 0.f) ? (x + log1pf(expf(-x))) : log1pf(expf(x));
}

// A-fragment (16x32 f16) per-lane K index for half element i:
// contiguous K = (lane/16)*16 + i (matches contiguous B loads)
__device__ __forceinline__ int a_k_idx(int i, int lane) {
    return ((lane >> 4) << 4) + i;
}

__device__ __forceinline__ v8f wmma_f16(v16h a, v16h b, v8f c) {
    return __builtin_amdgcn_wmma_f32_16x16x32_f16(false, a, false, b, (short)0, c,
                                                  false, false);
}

// -------------------------- threefry2x32 (JAX key scheme) for slot noise ---
__device__ __forceinline__ void tf_round(unsigned &x0, unsigned &x1, int r) {
    x0 += x1; x1 = (x1 << r) | (x1 >> (32 - r)); x1 ^= x0;
}
__device__ __forceinline__ void threefry2x32(unsigned k0, unsigned k1,
                                             unsigned x0, unsigned x1,
                                             unsigned &o0, unsigned &o1) {
    unsigned ks2 = k0 ^ k1 ^ 0x1BD11BDAu;
    x0 += k0; x1 += k1;
    tf_round(x0,x1,13); tf_round(x0,x1,15); tf_round(x0,x1,26); tf_round(x0,x1,6);
    x0 += k1; x1 += ks2 + 1u;
    tf_round(x0,x1,17); tf_round(x0,x1,29); tf_round(x0,x1,16); tf_round(x0,x1,24);
    x0 += ks2; x1 += k0 + 2u;
    tf_round(x0,x1,13); tf_round(x0,x1,15); tf_round(x0,x1,26); tf_round(x0,x1,6);
    x0 += k0; x1 += k1 + 3u;
    tf_round(x0,x1,17); tf_round(x0,x1,29); tf_round(x0,x1,16); tf_round(x0,x1,24);
    x0 += k1; x1 += ks2 + 4u;
    tf_round(x0,x1,13); tf_round(x0,x1,15); tf_round(x0,x1,26); tf_round(x0,x1,6);
    o0 = x0 + ks2; o1 = x1 + k0 + 5u;
}

// ------------------------------------------------------------ prep kernel ---
// Fold BN into scale/shift, transpose conv weights to [oc][k] f16 (k = tap*64+ic),
// transpose dense weights to [n][k] f16 for WMMA B-operands.
__global__ void prep_params(
    const float* w1, const float* w2, const float* wl1, const float* wl2,
    const float* g1, const float* b1, const float* m1, const float* v1,
    const float* g2, const float* b2, const float* m2, const float* v2,
    const float* bl1, const float* bl2,
    const float* wg1, const float* wg2, const float* wk, const float* wv,
    _Float16* wT1, _Float16* wT2, _Float16* wTl1, _Float16* wTl2,
    _Float16* wg1T, _Float16* wg2T, _Float16* wkT, _Float16* wvT,
    float* sc1, float* sh1, float* sc2, float* sh2,
    float* scl1, float* shl1, float* scl2, float* shl2) {
    int tid = blockIdx.x * blockDim.x + threadIdx.x;
    int nth = gridDim.x * blockDim.x;
    for (int idx = tid; idx < 36864; idx += nth) {
        int o = idx / 576, r = idx % 576, tap = r >> 6, ic = r & 63;
        int s = o * 576 + ic * 9 + tap;   // src OIHW (O,64,3,3)
        wT1[idx]  = (_Float16)w1[s];
        wT2[idx]  = (_Float16)w2[s];
        wTl1[idx] = (_Float16)wl1[s];
        wTl2[idx] = (_Float16)wl2[s];
    }
    for (int idx = tid; idx < 4096; idx += nth) {
        int n = idx >> 6, k = idx & 63;
        wg1T[idx] = (_Float16)wg1[k * 64 + n];
        wg2T[idx] = (_Float16)wg2[k * 64 + n];
        wkT[idx]  = (_Float16)wk[k * 64 + n];
        wvT[idx]  = (_Float16)wv[k * 64 + n];
    }
    for (int c = tid; c < 64; c += nth) {
        float s1 = g1[c] * rsqrtf(v1[c] + 1e-5f);
        sc1[c] = s1; sh1[c] = b1[c] - m1[c] * s1;
        float s2 = g2[c] * rsqrtf(v2[c] + 1e-5f);
        sc2[c] = s2; sh2[c] = b2[c] - m2[c] * s2;
        scl1[c] = 1.f; shl1[c] = bl1[c];
        scl2[c] = 1.f; shl2[c] = bl2[c];
    }
}

// ------------------------------------------ 1) patch embed + channel LN ----
// x (B,1,64,64) f32 -> f1 (B,64,64,64ch) f16 channels-last
__global__ void patch_embed_ln(const float* __restrict__ x,
                               const float* __restrict__ w,     // (64,1,3,3)
                               const float* __restrict__ bias,
                               const float* __restrict__ g,
                               const float* __restrict__ beta,
                               _Float16* __restrict__ out) {
    int p = blockIdx.x * blockDim.x + threadIdx.x;       // B*64*64 pixels
    int b = p >> 12, yx = p & 4095, y = yx >> 6, xq = yx & 63;
    const float* xb = x + (size_t)b * 4096;
    float tap[9];
    #pragma unroll
    for (int ky = 0; ky < 3; ky++)
        #pragma unroll
        for (int kx = 0; kx < 3; kx++) {
            int iy = y - 1 + ky, ix = xq - 1 + kx;
            tap[ky * 3 + kx] = (iy >= 0 && iy < 64 && ix >= 0 && ix < 64)
                                   ? xb[iy * 64 + ix] : 0.f;
        }
    float v[64];
    float mean = 0.f;
    for (int d = 0; d < 64; d++) {
        float acc = bias[d];
        const float* wr = w + d * 9;
        #pragma unroll
        for (int t = 0; t < 9; t++) acc += tap[t] * wr[t];
        v[d] = acc; mean += acc;
    }
    mean *= (1.f / 64.f);
    float var = 0.f;
    for (int d = 0; d < 64; d++) { float t = v[d] - mean; var += t * t; }
    float inv = rsqrtf(var * (1.f / 64.f) + 1e-5f);
    _Float16* o = out + (size_t)p * 64;
    for (int d = 0; d < 64; d++)
        o[d] = (_Float16)((v[d] - mean) * inv * g[d] + beta[d]);
}

// ------------------------- 2) 3x3 conv as WMMA GEMM + scale/shift + GELU ---
// in (B,Hin,Win,64) f16, wT (64oc x 576k) f16, out (B,Hout,Wout,64) f16
// Block: 128 threads = 4 waves; tile = 16 output pixels x 64 output channels.
__global__ void conv3x3_wmma(const _Float16* __restrict__ in,
                             const _Float16* __restrict__ wT,
                             const float* __restrict__ scale,
                             const float* __restrict__ shift,
                             _Float16* __restrict__ out,
                             int Hin, int Win, int Hout, int Wout,
                             int stride, int pad, int dil) {
    __shared__ _Float16 sA[16 * 576];            // 16 pixels x (9 taps * 64 ch)
    const int tid = threadIdx.x;
    const int HWo = Hout * Wout;
    const int p0 = blockIdx.x * 16;
    const int b = p0 / HWo;
    const int base = p0 % HWo;

    for (int idx = tid; idx < 16 * 576; idx += 128) {
        int m = idx / 576, k = idx % 576;
        int t = k >> 6, c = k & 63;
        int p = base + m;
        int oy = p / Wout, ox = p % Wout;
        int ky = t / 3, kx = t % 3;
        int iy = oy * stride - pad + ky * dil;
        int ix = ox * stride - pad + kx * dil;
        _Float16 vv = (_Float16)0.f;
        if (iy >= 0 && iy < Hin && ix >= 0 && ix < Win)
            vv = in[(((size_t)b * Hin + iy) * Win + ix) * 64 + c];
        sA[idx] = vv;
    }
    __syncthreads();

    const int lane = tid & 31, wave = tid >> 5;
    const int col = wave * 16 + (lane & 15);     // output channel for this lane
    const _Float16* bptr = wT + (size_t)col * 576 + ((lane >> 4) << 4);
    __builtin_prefetch(bptr, 0, 1);              // global_prefetch_b8
    v8f acc = {0.f, 0.f, 0.f, 0.f, 0.f, 0.f, 0.f, 0.f};
    for (int kc = 0; kc < 576; kc += 32) {
        v16h a;
        #pragma unroll
        for (int i = 0; i < 16; i++)
            a[i] = sA[(lane & 15) * 576 + kc + a_k_idx(i, lane)];
        v16h bv = *(const v16h*)(bptr + kc);     // 32B-aligned contiguous K
        acc = wmma_f16(a, bv, acc);
    }
    const float sc = scale[col], sh = shift[col];
    #pragma unroll
    for (int j = 0; j < 8; j++) {
        int m = j + ((lane >> 4) << 3);
        int p = base + m;
        int oy = p / Wout, ox = p % Wout;
        float vv = geluf(acc[j] * sc + sh);
        out[(((size_t)b * Hout + oy) * Wout + ox) * 64 + col] = (_Float16)vv;
    }
}

// ----------------------------- generic K=64 WMMA GEMM (dense projections) ---
// A (M x 64) f16 row-major, Bt (N x 64) f16, C = act(A@B + bias) to f32/f16.
__global__ void wmma_gemm64(const _Float16* __restrict__ A,
                            const _Float16* __restrict__ Bt,
                            const float* __restrict__ bias,
                            float* __restrict__ Cf, _Float16* __restrict__ Ch,
                            int N, int act) {
    __shared__ _Float16 sA[16 * 64];
    const int tid = threadIdx.x;
    const int mbase = blockIdx.x * 16;
    for (int idx = tid; idx < 1024; idx += 128)
        sA[idx] = A[(size_t)mbase * 64 + idx];
    __syncthreads();
    const int lane = tid & 31, wave = tid >> 5;
    const int nb = wave * 16;
    if (nb >= N) return;
    const int col = nb + (lane & 15);
    const _Float16* bptr = Bt + (size_t)col * 64 + ((lane >> 4) << 4);
    v8f acc = {0.f, 0.f, 0.f, 0.f, 0.f, 0.f, 0.f, 0.f};
    #pragma unroll
    for (int kc = 0; kc < 64; kc += 32) {
        v16h a;
        #pragma unroll
        for (int i = 0; i < 16; i++)
            a[i] = sA[(lane & 15) * 64 + kc + a_k_idx(i, lane)];
        v16h bv = *(const v16h*)(bptr + kc);
        acc = wmma_f16(a, bv, acc);
    }
    const float bs = bias ? bias[col] : 0.f;
    #pragma unroll
    for (int j = 0; j < 8; j++) {
        int m = mbase + j + ((lane >> 4) << 3);
        float vv = acc[j] + bs;
        if (act == 1) vv = geluf(vv);
        if (Cf) Cf[(size_t)m * N + col] = vv;
        if (Ch) Ch[(size_t)m * N + col] = (_Float16)vv;
    }
}

// ------------------------------------- 3a) global mean over 16x16 pixels ---
__global__ void global_mean(const _Float16* __restrict__ f3,
                            _Float16* __restrict__ gm) {
    int b = blockIdx.x, c = threadIdx.x;                 // 64 threads
    const _Float16* p = f3 + (size_t)b * 16384 + c;
    float s = 0.f;
    for (int i = 0; i < 256; i++) s += (float)p[(size_t)i * 64];
    gm[b * 64 + c] = (_Float16)(s * (1.f / 256.f));
}

// ------ 3b) alpha gate + fuse + feature LayerNorm (one thread per pixel) ---
__global__ void fuse_alpha_ln(const _Float16* __restrict__ loc2,
                              const float* __restrict__ g2,
                              const float* __restrict__ wa,  // (64 x 128)
                              const float* __restrict__ ba,
                              const float* __restrict__ lg,
                              const float* __restrict__ lb,
                              _Float16* __restrict__ lnfeats) {
    int b = blockIdx.x, n = threadIdx.x;                 // 256 pixels
    __shared__ float gs[64];
    if (n < 64) gs[n] = g2[b * 64 + n];
    __syncthreads();
    const _Float16* lp = loc2 + ((size_t)b * 256 + n) * 64;
    float l[64];
    for (int c = 0; c < 64; c++) l[c] = (float)lp[c];
    float f[64];
    float mean = 0.f;
    for (int d = 0; d < 64; d++) {
        const float* wr = wa + d * 128;
        float acc = ba[d];
        for (int c = 0; c < 64; c++) acc += wr[c] * l[c] + wr[64 + c] * gs[c];
        float al = sigm(acc);
        float fu = al * l[d] + (1.f - al) * gs[d];
        f[d] = fu; mean += fu;
    }
    mean *= (1.f / 64.f);
    float var = 0.f;
    for (int d = 0; d < 64; d++) { float t = f[d] - mean; var += t * t; }
    float inv = rsqrtf(var * (1.f / 64.f) + 1e-5f);
    _Float16* op = lnfeats + ((size_t)b * 256 + n) * 64;
    for (int d = 0; d < 64; d++)
        op[d] = (_Float16)((f[d] - mean) * inv * lg[d] + lb[d]);
}

// ----- shared-memory LayerNorm over D=64 per slot (4 slots, 256 threads) ---
__device__ __forceinline__ void ln64(const float* src, float* dst,
                                     const float* g, const float* bb,
                                     float* stat, int tid, int s, int d) {
    __syncthreads();
    if (tid < 4) {
        float m = 0.f;
        for (int k = 0; k < 64; k++) m += src[tid * 64 + k];
        m *= (1.f / 64.f);
        float v = 0.f;
        for (int k = 0; k < 64; k++) { float t = src[tid * 64 + k] - m; v += t * t; }
        stat[tid * 2] = m;
        stat[tid * 2 + 1] = rsqrtf(v * (1.f / 64.f) + 1e-5f);
    }
    __syncthreads();
    dst[tid] = (src[tid] - stat[s * 2]) * stat[s * 2 + 1] * g[d] + bb[d];
    __syncthreads();
}

// -------------- 4+5) slot attention (3 iters) + GRU + MLP + SlotMamba ------
// one workgroup (256 threads) per batch element; k panel cached in LDS f16
__global__ void slot_attn_mamba(
    const _Float16* __restrict__ kmat, const float* __restrict__ vmat,
    const float* mu, const float* log_sigma,
    const float* lns_g, const float* lns_b, const float* lnm_g, const float* lnm_b,
    const float* wq, const float* wi, const float* bi,
    const float* wh, const float* bh,
    const float* w1, const float* b1, const float* w2, const float* b2,
    const float* ww, const float* bw,
    const float* smln_g, const float* smln_b, const float* w_in,
    const float* w_conv, const float* b_conv, const float* w_x,
    const float* w_dt, const float* b_dt, const float* A_log, const float* Dp,
    const float* w_out, float* __restrict__ outp) {
    const int b = blockIdx.x, tid = threadIdx.x;
    const int s = tid >> 6, d = tid & 63;

    __shared__ _Float16 sk[256 * 64];      // 32 KB
    __shared__ float attnw[4 * 256];
    __shared__ float qv[4 * 64];
    __shared__ float slots[4 * 64];
    __shared__ float prev[4 * 64];
    __shared__ float lnb[4 * 64];
    __shared__ float upd[4 * 64];
    __shared__ float h1[4 * 64];
    __shared__ float xmv[4 * 64];
    __shared__ float zgv[4 * 64];
    __shared__ float ybuf[4 * 64];
    __shared__ float xdbl[4 * 36];
    __shared__ float stat[8];
    __shared__ float colsum[4];

    for (int idx = tid; idx < 16384; idx += 256)
        sk[idx] = kmat[(size_t)b * 16384 + idx];

    {   // slots = mu + exp(log_sigma) * N(key(42))   [JAX threefry2x32 scheme]
        unsigned i = (unsigned)(b * 256 + tid);
        unsigned x0, x1; int sel;
        if (i < 16384u) { x0 = i; x1 = i + 16384u; sel = 0; }
        else            { x0 = i - 16384u; x1 = i; sel = 1; }
        unsigned o0, o1;
        threefry2x32(0u, 42u, x0, x1, o0, o1);
        unsigned bits = sel ? o1 : o0;
        float u01 = __uint_as_float((bits >> 9) | 0x3f800000u) - 1.0f;
        const float lo = -0.99999994f;
        float u = u01 * (1.0f - lo) + lo;
        if (u < lo) u = lo;
        float nz = 1.4142135623730951f * erfinvf(u);
        slots[tid] = mu[d] + expf(log_sigma[d]) * nz;
    }
    __syncthreads();

    const float scale = 0.125f;  // D^-0.5
    for (int it = 0; it < 3; it++) {
        prev[tid] = slots[tid];
        ln64(slots, lnb, lns_g, lns_b, stat, tid, s, d);
        {   // q = LN(slots) @ wq
            float acc = 0.f;
            for (int k = 0; k < 64; k++) acc += lnb[s * 64 + k] * wq[k * 64 + d];
            qv[tid] = acc;
        }
        if (tid < 4) colsum[tid] = 0.f;
        __syncthreads();
        {   // attention per feature n = tid (softmax over 4 slots)
            float lg[4];
            float mx = -1e30f;
            const _Float16* kr = sk + tid * 64;
            for (int ss = 0; ss < 4; ss++) {
                float acc = 0.f;
                for (int d2 = 0; d2 < 64; d2++)
                    acc += (float)kr[d2] * qv[ss * 64 + d2];
                lg[ss] = acc * scale;
                mx = fmaxf(mx, lg[ss]);
            }
            float sum = 0.f;
            for (int ss = 0; ss < 4; ss++) { lg[ss] = expf(lg[ss] - mx); sum += lg[ss]; }
            float inv = 1.f / sum;
            for (int ss = 0; ss < 4; ss++) {
                float a = lg[ss] * inv;
                attnw[ss * 256 + tid] = a;
                atomicAdd(&colsum[ss], a);
            }
        }
        __syncthreads();
        {   // updates = (attn / colsum) @ v
            float inv = 1.f / (colsum[s] + 1e-8f);
            float acc = 0.f;
            const float* vp = vmat + (size_t)b * 16384 + d;
            const float* aw = attnw + s * 256;
            for (int n2 = 0; n2 < 256; n2++) acc += aw[n2] * vp[(size_t)n2 * 64];
            upd[tid] = acc * inv;
        }
        __syncthreads();
        {   // GRUCell (torch convention)
            float gir = bi[d], giz = bi[64 + d], gin = bi[128 + d];
            float ghr = bh[d], ghz = bh[64 + d], ghn = bh[128 + d];
            for (int k = 0; k < 64; k++) {
                float u = upd[s * 64 + k], pz = prev[s * 64 + k];
                const float* wir = wi + k * 192;
                const float* whr = wh + k * 192;
                gir += u * wir[d]; giz += u * wir[64 + d]; gin += u * wir[128 + d];
                ghr += pz * whr[d]; ghz += pz * whr[64 + d]; ghn += pz * whr[128 + d];
            }
            float r = sigm(gir + ghr), z = sigm(giz + ghz);
            float nn = tanhf(gin + r * ghn);
            slots[tid] = (1.f - z) * nn + z * prev[s * 64 + d];
        }
        // residual MLP
        ln64(slots, lnb, lnm_g, lnm_b, stat, tid, s, d);
        {
            float acc = b1[d];
            for (int k = 0; k < 64; k++) acc += lnb[s * 64 + k] * w1[k * 64 + d];
            h1[tid] = fmaxf(acc, 0.f);
        }
        __syncthreads();
        {
            float acc = b2[d];
            for (int j = 0; j < 64; j++) acc += h1[s * 64 + j] * w2[j * 64 + d];
            slots[tid] += acc;
        }
        __syncthreads();
    }

    // slot weights
    if (tid < 4) {
        float acc = bw[0];
        for (int d2 = 0; d2 < 64; d2++) acc += slots[tid * 64 + d2] * ww[d2];
        outp[32768 + b * 4 + tid] = sigm(acc);
    }

    // -------- SlotMamba over the K=4 slot sequence --------
    prev[tid] = slots[tid];                       // residual
    ln64(slots, lnb, smln_g, smln_b, stat, tid, s, d);
    {   // [xm | zg] = h @ w_in  (D x 2D)
        float xm = 0.f, zg = 0.f;
        for (int k = 0; k < 64; k++) {
            float hh = lnb[s * 64 + k];
            const float* wr = w_in + k * 128;
            xm += hh * wr[d]; zg += hh * wr[64 + d];
        }
        xmv[tid] = xm; zgv[tid] = zg;
    }
    __syncthreads();
    {   // causal depthwise conv (DCONV=4, left pad 3) + SiLU
        float acc = b_conv[d];
        #pragma unroll
        for (int t = 0; t < 4; t++) {
            int sp = s - 3 + t;
            if (sp >= 0) acc += xmv[sp * 64 + d] * w_conv[d * 4 + t];
        }
        float sx = acc * sigm(acc);
        __syncthreads();
        xmv[tid] = sx;
    }
    __syncthreads();
    if (tid < 144) {                              // x_dbl = xm @ w_x (D x 36)
        int ss = tid / 36, c = tid % 36;
        float acc = 0.f;
        for (int k = 0; k < 64; k++) acc += xmv[ss * 64 + k] * w_x[k * 36 + c];
        xdbl[tid] = acc;
    }
    __syncthreads();
    {   // dt = softplus(dt_in @ w_dt + b_dt)   -> reuse lnb
        float acc = b_dt[d];
        #pragma unroll
        for (int r2 = 0; r2 < 4; r2++) acc += xdbl[s * 36 + r2] * w_dt[r2 * 64 + d];
        lnb[tid] = softplusf(acc);
    }
    __syncthreads();
    if (tid < 64) {                               // selective-scan, channel = tid
        float hs[16];
        #pragma unroll
        for (int j = 0; j < 16; j++) hs[j] = 0.f;
        for (int ss = 0; ss < 4; ss++) {
            float dtc = lnb[ss * 64 + tid];
            float xv = xmv[ss * 64 + tid];
            float y = 0.f;
            #pragma unroll
            for (int j = 0; j < 16; j++) {
                float Aj = -expf(A_log[tid * 16 + j]);
                float dA = expf(dtc * Aj);
                hs[j] = dA * hs[j] + dtc * xdbl[ss * 36 + 4 + j] * xv;  // B cols 4..19
                y += hs[j] * xdbl[ss * 36 + 20 + j];                    // C cols 20..35
            }
            ybuf[ss * 64 + tid] = y;
        }
    }
    __syncthreads();
    {   // gate: y = (ys + Dp*xm) * silu(zg)  -> reuse h1
        float y = ybuf[s * 64 + d] + Dp[d] * xmv[s * 64 + d];
        float zg = zgv[s * 64 + d];
        h1[tid] = y * zg * sigm(zg);
    }
    __syncthreads();
    {   // out = y @ w_out + residual
        float acc = prev[s * 64 + d];
        for (int j = 0; j < 64; j++) acc += h1[s * 64 + j] * w_out[j * 64 + d];
        outp[b * 256 + s * 64 + d] = acc;
    }
}

// =========================================================== launcher ======
extern "C" void kernel_launch(void* const* d_in, const int* in_sizes, int n_in,
                              void* d_out, int out_size, void* d_ws, size_t ws_size,
                              hipStream_t stream) {
    (void)in_sizes; (void)n_in; (void)out_size; (void)ws_size;
#define IN(i) ((const float*)d_in[(i)])
    const float* x       = IN(0);
    const float* pe_w    = IN(1);  const float* pe_bias = IN(2);
    const float* pe_g    = IN(3);  const float* pe_beta = IN(4);
    const float* ds_w1   = IN(5);  const float* ds_g1 = IN(6);
    const float* ds_b1   = IN(7);  const float* ds_m1 = IN(8);  const float* ds_v1 = IN(9);
    const float* ds_w2   = IN(10); const float* ds_g2 = IN(11);
    const float* ds_b2   = IN(12); const float* ds_m2 = IN(13); const float* ds_v2 = IN(14);
    const float* db_wl1  = IN(15); const float* db_bl1 = IN(16);
    const float* db_wl2  = IN(17); const float* db_bl2 = IN(18);
    const float* db_wg1  = IN(19); const float* db_bg1 = IN(20);
    const float* db_wg2  = IN(21); const float* db_bg2 = IN(22);
    const float* db_wa   = IN(23); const float* db_ba  = IN(24);
    const float* sa_mu   = IN(25); const float* sa_ls  = IN(26);
    const float* lni_g   = IN(27); const float* lni_b  = IN(28);
    const float* lns_g   = IN(29); const float* lns_b  = IN(30);
    const float* lnm_g   = IN(31); const float* lnm_b  = IN(32);
    const float* sa_wq   = IN(33); const float* sa_wk  = IN(34); const float* sa_wv = IN(35);
    const float* sa_wi   = IN(36); const float* sa_bi  = IN(37);
    const float* sa_wh   = IN(38); const float* sa_bh  = IN(39);
    const float* sa_w1   = IN(40); const float* sa_b1  = IN(41);
    const float* sa_w2   = IN(42); const float* sa_b2  = IN(43);
    const float* sa_ww   = IN(44); const float* sa_bw  = IN(45);
    const float* sm_lng  = IN(46); const float* sm_lnb = IN(47);
    const float* sm_win  = IN(48); const float* sm_wc  = IN(49); const float* sm_bc = IN(50);
    const float* sm_wx   = IN(51); const float* sm_wdt = IN(52); const float* sm_bdt = IN(53);
    const float* sm_Alog = IN(54); const float* sm_Dp  = IN(55); const float* sm_wout = IN(56);
#undef IN

    char* ws = (char*)d_ws;
    size_t off = 0;
    auto carve = [&](size_t bytes) -> void* {
        void* p = ws + off;
        off += (bytes + 255) & ~(size_t)255;
        return p;
    };
    _Float16* f1      = (_Float16*)carve(67108864);  // (128,64,64,64) f16
    _Float16* f2      = (_Float16*)carve(16777216);  // (128,32,32,64)
    _Float16* f3      = (_Float16*)carve(4194304);   // (128,16,16,64)
    _Float16* loc1    = (_Float16*)carve(4194304);
    _Float16* loc2    = (_Float16*)carve(4194304);
    _Float16* lnfeats = (_Float16*)carve(4194304);   // LN(feats) (B,256,64)
    _Float16* kmat    = (_Float16*)carve(4194304);   // k (B,256,64) f16
    float*    vmat    = (float*)carve(8388608);      // v (B,256,64) f32
    _Float16* gmean   = (_Float16*)carve(16384);
    _Float16* g1h     = (_Float16*)carve(16384);
    float*    g2f     = (float*)carve(32768);
    _Float16* wT1  = (_Float16*)carve(73728);
    _Float16* wT2  = (_Float16*)carve(73728);
    _Float16* wTl1 = (_Float16*)carve(73728);
    _Float16* wTl2 = (_Float16*)carve(73728);
    _Float16* wg1T = (_Float16*)carve(8192);
    _Float16* wg2T = (_Float16*)carve(8192);
    _Float16* wkT  = (_Float16*)carve(8192);
    _Float16* wvT  = (_Float16*)carve(8192);
    float* sc1  = (float*)carve(256); float* sh1  = (float*)carve(256);
    float* sc2  = (float*)carve(256); float* sh2  = (float*)carve(256);
    float* scl1 = (float*)carve(256); float* shl1 = (float*)carve(256);
    float* scl2 = (float*)carve(256); float* shl2 = (float*)carve(256);

    prep_params<<<64, 256, 0, stream>>>(
        ds_w1, ds_w2, db_wl1, db_wl2,
        ds_g1, ds_b1, ds_m1, ds_v1, ds_g2, ds_b2, ds_m2, ds_v2,
        db_bl1, db_bl2, db_wg1, db_wg2, sa_wk, sa_wv,
        wT1, wT2, wTl1, wTl2, wg1T, wg2T, wkT, wvT,
        sc1, sh1, sc2, sh2, scl1, shl1, scl2, shl2);

    patch_embed_ln<<<2048, 256, 0, stream>>>(x, pe_w, pe_bias, pe_g, pe_beta, f1);

    conv3x3_wmma<<<8192, 128, 0, stream>>>(f1, wT1, sc1, sh1, f2, 64, 64, 32, 32, 2, 1, 1);
    conv3x3_wmma<<<2048, 128, 0, stream>>>(f2, wT2, sc2, sh2, f3, 32, 32, 16, 16, 2, 1, 1);
    conv3x3_wmma<<<2048, 128, 0, stream>>>(f3, wTl1, scl1, shl1, loc1, 16, 16, 16, 16, 1, 1, 1);
    conv3x3_wmma<<<2048, 128, 0, stream>>>(loc1, wTl2, scl2, shl2, loc2, 16, 16, 16, 16, 1, 2, 2);

    global_mean<<<128, 64, 0, stream>>>(f3, gmean);
    wmma_gemm64<<<8, 128, 0, stream>>>(gmean, wg1T, db_bg1, nullptr, g1h, 64, 1);
    wmma_gemm64<<<8, 128, 0, stream>>>(g1h, wg2T, db_bg2, g2f, nullptr, 64, 0);

    fuse_alpha_ln<<<128, 256, 0, stream>>>(loc2, g2f, db_wa, db_ba, lni_g, lni_b, lnfeats);

    wmma_gemm64<<<2048, 128, 0, stream>>>(lnfeats, wkT, nullptr, nullptr, kmat, 64, 0);
    wmma_gemm64<<<2048, 128, 0, stream>>>(lnfeats, wvT, nullptr, vmat, nullptr, 64, 0);

    slot_attn_mamba<<<128, 256, 0, stream>>>(
        kmat, vmat, sa_mu, sa_ls, lns_g, lns_b, lnm_g, lnm_b,
        sa_wq, sa_wi, sa_bi, sa_wh, sa_bh, sa_w1, sa_b1, sa_w2, sa_b2,
        sa_ww, sa_bw, sm_lng, sm_lnb, sm_win, sm_wc, sm_bc, sm_wx,
        sm_wdt, sm_bdt, sm_Alog, sm_Dp, sm_wout, (float*)d_out);
}
